// MACRO_VRNN_37039797961347
// MI455X (gfx1250) — matrix operand order repaired and checked by reading, not compiled, not run on hardware
//
#include <hip/hip_runtime.h>

#define A_ 10
#define T_ 65
#define B_ 128
#define XD_ 2
#define YD_ 20
#define ZD_ 64
#define HD_ 256
#define MD_ 90
#define RM_ 256
#define LOG2PI_ 1.8378770664093453f

typedef __attribute__((ext_vector_type(16))) __bf16        bf16x16;
typedef __attribute__((ext_vector_type(8)))  float         f32x8;
typedef __attribute__((ext_vector_type(4)))  unsigned int  u32x4;

union FragU { bf16x16 v; u32x4 q[2]; };

// global (addrspace 1) pointer types so weight/bias loads lower to
// global_load_* (LOADcnt only) instead of flat_load_* (LOADcnt+DScnt).
typedef const __attribute__((address_space(1))) unsigned short* gus_p;
typedef const __attribute__((address_space(1))) u32x4*          gq_p;
typedef const __attribute__((address_space(1))) float*          gf_p;
typedef const __attribute__((address_space(1))) int*            gi_p;

__device__ inline float ldg_f(const float* p) { return *(gf_p)p; }
__device__ inline int   ldg_i(const int*   p) { return *(gi_p)p; }

// ---- workspace layout (bf16 elements), per agent ----
static const long long OFF_ENC_WX  = 0;        // [256][32]
static const long long OFF_ENC_WH  = 8192;     // [256][256]
static const long long OFF_ENC_W2  = 73728;    // [256][256]
static const long long OFF_ENC_WM  = 139264;   // [64][256]
static const long long OFF_ENC_WS  = 155648;   // [64][256]
static const long long OFF_PRI_WH  = 172032;   // [256][256]
static const long long OFF_PRI_W2  = 237568;   // [256][256]
static const long long OFF_PRI_WM  = 303104;   // [64][256]
static const long long OFF_PRI_WS  = 319488;   // [64][256]
static const long long OFF_DEC_WY  = 335872;   // [256][32]
static const long long OFF_DEC_WZ  = 344064;   // [256][64]
static const long long OFF_DEC_WH  = 360448;   // [256][256]
static const long long OFF_DEC_W2  = 425984;   // [256][256]
static const long long OFF_DEC_WMH = 491520;   // [16][256]
static const long long OFF_DEC_WSH = 495616;   // [16][256]
static const long long OFF_GRU_WX  = 499712;   // [768][32]
static const long long OFF_GRU_WZ  = 524288;   // [768][64]
static const long long OFF_GRU_WH  = 573440;   // [768][256]
static const long long AGW_        = 770048;   // per-agent stride

__device__ __host__ inline unsigned short f2bf(float f) {
  union { float f; unsigned int u; } x; x.f = f;
  unsigned int r = x.u + 0x7FFFu + ((x.u >> 16) & 1u);
  return (unsigned short)(r >> 16);
}

__device__ inline float softplusf(float x) {
  return (x > 20.f) ? x : log1pf(expf(x));
}
__device__ inline float sigmoidf_(float x) {
  return 1.f / (1.f + expf(-x));
}

// A-frag (16-bit 16x32): lanes 0-15 -> K {0..7,16..23}; lanes 16-31 -> K {8..15,24..31}
// B-frag (16-bit 32x16): lanes 0-15 -> K {0..7,16..23}; lanes 16-31 -> K {8..15,24..31}

// single-tile accumulate
__device__ inline void mm_acc(f32x8& c, const unsigned short* aL, int aK,
                              const unsigned short* wT, int wK,
                              int ntile, int kchunks, int lane) {
  const int ah = (lane & 16) ? 8 : 0;
  const int bh = (lane & 16) ? 8 : 0;
  const unsigned short* ap = aL + (lane & 15) * aK + ah;
  gus_p bp = (gus_p)wT + (long long)(ntile * 16 + (lane & 15)) * wK + bh;
  #pragma unroll
  for (int kc = 0; kc < kchunks; ++kc) {
    FragU a, b;
    a.q[0] = *(const u32x4*)(ap);
    a.q[1] = *(const u32x4*)(ap + 16);
    b.q[0] = *(gq_p)(bp);
    b.q[1] = *(gq_p)(bp + 16);
    c = __builtin_amdgcn_wmma_f32_16x16x32_bf16(false, a.v, false, b.v,
                                                (short)0, c, false, false);
    ap += 32; bp += 32;
  }
}

// two N-tiles sharing one A-fragment load
__device__ inline void mm_acc2(f32x8& c0, f32x8& c1,
                               const unsigned short* aL, int aK,
                               const unsigned short* wT, int wK,
                               int nt0, int nt1, int kchunks, int lane) {
  const int ah = (lane & 16) ? 8 : 0;
  const int bh = (lane & 16) ? 8 : 0;
  const unsigned short* ap = aL + (lane & 15) * aK + ah;
  gus_p bp0 = (gus_p)wT + (long long)(nt0 * 16 + (lane & 15)) * wK + bh;
  gus_p bp1 = (gus_p)wT + (long long)(nt1 * 16 + (lane & 15)) * wK + bh;
  #pragma unroll
  for (int kc = 0; kc < kchunks; ++kc) {
    FragU a, b0, b1;
    a.q[0]  = *(const u32x4*)(ap);
    a.q[1]  = *(const u32x4*)(ap + 16);
    b0.q[0] = *(gq_p)(bp0);  b0.q[1] = *(gq_p)(bp0 + 16);
    b1.q[0] = *(gq_p)(bp1);  b1.q[1] = *(gq_p)(bp1 + 16);
    c0 = __builtin_amdgcn_wmma_f32_16x16x32_bf16(false, a.v, false, b0.v,
                                                 (short)0, c0, false, false);
    c1 = __builtin_amdgcn_wmma_f32_16x16x32_bf16(false, a.v, false, b1.v,
                                                 (short)0, c1, false, false);
    ap += 32; bp0 += 32; bp1 += 32;
  }
}

// three N-tiles (GRU r/u/n gates) sharing one A-fragment load
__device__ inline void mm_acc3(f32x8& c0, f32x8& c1, f32x8& c2,
                               const unsigned short* aL, int aK,
                               const unsigned short* wT, int wK,
                               int nt0, int nt1, int nt2, int kchunks, int lane) {
  const int ah = (lane & 16) ? 8 : 0;
  const int bh = (lane & 16) ? 8 : 0;
  const unsigned short* ap = aL + (lane & 15) * aK + ah;
  gus_p bp0 = (gus_p)wT + (long long)(nt0 * 16 + (lane & 15)) * wK + bh;
  gus_p bp1 = (gus_p)wT + (long long)(nt1 * 16 + (lane & 15)) * wK + bh;
  gus_p bp2 = (gus_p)wT + (long long)(nt2 * 16 + (lane & 15)) * wK + bh;
  #pragma unroll
  for (int kc = 0; kc < kchunks; ++kc) {
    FragU a, b0, b1, b2;
    a.q[0]  = *(const u32x4*)(ap);
    a.q[1]  = *(const u32x4*)(ap + 16);
    b0.q[0] = *(gq_p)(bp0);  b0.q[1] = *(gq_p)(bp0 + 16);
    b1.q[0] = *(gq_p)(bp1);  b1.q[1] = *(gq_p)(bp1 + 16);
    b2.q[0] = *(gq_p)(bp2);  b2.q[1] = *(gq_p)(bp2 + 16);
    c0 = __builtin_amdgcn_wmma_f32_16x16x32_bf16(false, a.v, false, b0.v,
                                                 (short)0, c0, false, false);
    c1 = __builtin_amdgcn_wmma_f32_16x16x32_bf16(false, a.v, false, b1.v,
                                                 (short)0, c1, false, false);
    c2 = __builtin_amdgcn_wmma_f32_16x16x32_bf16(false, a.v, false, b2.v,
                                                 (short)0, c2, false, false);
    ap += 32; bp0 += 32; bp1 += 32; bp2 += 32;
  }
}

// epilogue: c + one-hot gather row + bias, relu, -> bf16 LDS
__device__ inline void epi_gather_relu(const f32x8& c, int nt,
                                       const float* g, const float* b1,
                                       const int* sidx, unsigned short* Aout,
                                       int mb, int nl) {
  int n = nt * 16 + nl;
  float bn = ldg_f(b1 + n);
  #pragma unroll
  for (int v = 0; v < 8; ++v) {
    int m = mb + v;
    float val = c[v] + ldg_f(g + sidx[m] * 256 + n) + bn;
    Aout[m * 256 + n] = f2bf(fmaxf(val, 0.f));
  }
}

__device__ inline void epi_bias_relu(const f32x8& c, int nt, const float* b,
                                     unsigned short* Aout, int mb, int nl) {
  int n = nt * 16 + nl;
  float bn = ldg_f(b + n);
  #pragma unroll
  for (int v = 0; v < 8; ++v)
    Aout[(mb + v) * 256 + n] = f2bf(fmaxf(c[v] + bn, 0.f));
}

__device__ inline void layer2_relu(const unsigned short* Ain, const unsigned short* Wt,
                                   const float* bias, unsigned short* Aout,
                                   int wave, int lane, int mb, int nl) {
  f32x8 c0 = {}, c1 = {};
  int nt0 = wave * 2, nt1 = wave * 2 + 1;
  mm_acc2(c0, c1, Ain, 256, Wt, 256, nt0, nt1, 8, lane);
  epi_bias_relu(c0, nt0, bias, Aout, mb, nl);
  epi_bias_relu(c1, nt1, bias, Aout, mb, nl);
}

__device__ inline void head64(const unsigned short* Ain, const unsigned short* Wt,
                              const float* bias, float* Fout, bool sp,
                              int nt, int lane, int mb, int nl) {
  f32x8 c = {};
  mm_acc(c, Ain, 256, Wt, 256, nt, 8, lane);
  int n = nt * 16 + nl;
  float bn = ldg_f(bias + n);
  #pragma unroll
  for (int v = 0; v < 8; ++v) {
    float val = c[v] + bn;
    if (sp) val = softplusf(val);
    Fout[(mb + v) * 64 + n] = val;
  }
}

__device__ inline void head2(const unsigned short* Ain, const unsigned short* Wt,
                             const float* bias2, float* Fout,
                             int lane, int mb, int nl) {
  f32x8 c = {};
  mm_acc(c, Ain, 256, Wt, 256, 0, 8, lane);
  float bn = (nl < 2) ? ldg_f(bias2 + nl) : 0.f;
  #pragma unroll
  for (int v = 0; v < 8; ++v)
    Fout[(mb + v) * 16 + nl] = c[v] + bn;
}

struct Params {
  const float* data; const int* mac; const float* eps;
  const float* encW1; const float* encB1; const float* encB2;
  const float* encmB; const float* encsB;
  const float* priW1; const float* priB1; const float* priB2;
  const float* primB; const float* prisB;
  const float* decW1; const float* decB1; const float* decB2;
  const float* decmB; const float* decsB;
  const float* gruBih; const float* gruBhh;
  const unsigned short* W; float* out;
};

__global__ void __launch_bounds__(256) vrnn_persistent(Params p) {
  const int a   = blockIdx.y;
  const int b0  = blockIdx.x * 16;
  const int tid = threadIdx.x;
  const int lane = tid & 31;
  const int wave = tid >> 5;
  const int mb   = (lane & 16) ? 8 : 0;
  const int nl   = lane & 15;

  __shared__ __align__(16) unsigned short s_h [16 * 256];
  __shared__ __align__(16) unsigned short s_a1[16 * 256];
  __shared__ __align__(16) unsigned short s_a2[16 * 256];
  __shared__ __align__(16) unsigned short s_x [16 * 32];
  __shared__ __align__(16) unsigned short s_y [16 * 32];
  __shared__ __align__(16) unsigned short s_z [16 * 64];
  __shared__ float s_em[16 * 64], s_es[16 * 64], s_pm[16 * 64], s_ps[16 * 64];
  __shared__ float s_dm[16 * 16], s_ds[16 * 16];
  __shared__ float s_xf[16 * 2];
  __shared__ int   s_idx[16];
  __shared__ float s_red[2];

  const unsigned short* Wag = p.W + (long long)a * AGW_;

  for (int i = tid; i < 16 * 256; i += 256) s_h[i] = 0;
  for (int i = tid; i < 16 * 32;  i += 256) { s_x[i] = 0; s_y[i] = 0; }
  if (tid < 2) s_red[tid] = 0.f;
  float hreg[2][8];
  #pragma unroll
  for (int i = 0; i < 2; ++i)
    #pragma unroll
    for (int v = 0; v < 8; ++v) hreg[i][v] = 0.f;
  float kl_acc = 0.f, rc_acc = 0.f;
  __syncthreads();

  for (int t = 0; t < T_ - 1; ++t) {
    // ---- prologue: x_t, y_t, macro idx ----
    if (tid < 16) s_idx[tid] = ldg_i(p.mac + ((long long)t * B_ + b0 + tid) * A_ + a);
    if (tid < 32) {
      int m = tid >> 1, j = tid & 1;
      float v = ldg_f(p.data + ((long long)(t + 1) * B_ + b0 + m) * YD_ + a * XD_ + j);
      s_xf[m * 2 + j] = v;
      s_x[m * 32 + j] = f2bf(v);
    }
    for (int i = tid; i < 16 * YD_; i += 256) {
      int m = i / YD_, j = i % YD_;
      s_y[m * 32 + j] = f2bf(ldg_f(p.data + ((long long)t * B_ + b0 + m) * YD_ + j));
    }
    __syncthreads();

    // ================= ENCODER =================
    {
      const float* g  = p.encW1 + ((long long)a * 348 + 2) * 256;   // one-hot rows
      const float* b1 = p.encB1 + a * 256;
      int nt0 = wave * 2, nt1 = wave * 2 + 1;
      f32x8 c0 = {}, c1 = {};
      mm_acc2(c0, c1, s_x, 32,  Wag + OFF_ENC_WX, 32,  nt0, nt1, 1, lane);
      mm_acc2(c0, c1, s_h, 256, Wag + OFF_ENC_WH, 256, nt0, nt1, 8, lane);
      epi_gather_relu(c0, nt0, g, b1, s_idx, s_a1, mb, nl);
      epi_gather_relu(c1, nt1, g, b1, s_idx, s_a1, mb, nl);
    }
    __syncthreads();
    layer2_relu(s_a1, Wag + OFF_ENC_W2, p.encB2 + a * 256, s_a2, wave, lane, mb, nl);
    __syncthreads();
    if (wave < 4) head64(s_a2, Wag + OFF_ENC_WM, p.encmB + a * 64, s_em, false, wave,     lane, mb, nl);
    else          head64(s_a2, Wag + OFF_ENC_WS, p.encsB + a * 64, s_es, true,  wave - 4, lane, mb, nl);
    __syncthreads();

    // ================= PRIOR =================
    {
      const float* g  = p.priW1 + (long long)a * 346 * 256;         // one-hot rows 0..89
      const float* b1 = p.priB1 + a * 256;
      int nt0 = wave * 2, nt1 = wave * 2 + 1;
      f32x8 c0 = {}, c1 = {};
      mm_acc2(c0, c1, s_h, 256, Wag + OFF_PRI_WH, 256, nt0, nt1, 8, lane);
      epi_gather_relu(c0, nt0, g, b1, s_idx, s_a1, mb, nl);
      epi_gather_relu(c1, nt1, g, b1, s_idx, s_a1, mb, nl);
    }
    __syncthreads();
    layer2_relu(s_a1, Wag + OFF_PRI_W2, p.priB2 + a * 256, s_a2, wave, lane, mb, nl);
    __syncthreads();
    if (wave < 4) head64(s_a2, Wag + OFF_PRI_WM, p.primB + a * 64, s_pm, false, wave,     lane, mb, nl);
    else          head64(s_a2, Wag + OFF_PRI_WS, p.prisB + a * 64, s_ps, true,  wave - 4, lane, mb, nl);
    __syncthreads();

    // ---- z = mu + eps*std ; KL accumulation ----
    for (int i = tid; i < 16 * 64; i += 256) {
      int m = i >> 6, k = i & 63;
      float em = s_em[i], es = s_es[i], pm = s_pm[i], ps = s_ps[i];
      float e = ldg_f(p.eps + ((((long long)t * A_ + a) * B_) + b0 + m) * 64 + k);
      s_z[m * 64 + k] = f2bf(em + e * es);
      float dm = em - pm;
      kl_acc += 0.5f * (2.f * logf(ps) - 2.f * logf(es) + (es * es + dm * dm) / (ps * ps) - 1.f);
    }
    __syncthreads();

    // ================= DECODER =================
    {
      const float* g  = p.decW1 + ((long long)a * 430 + 20) * 256;  // one-hot rows
      const float* b1 = p.decB1 + a * 256;
      int nt0 = wave * 2, nt1 = wave * 2 + 1;
      f32x8 c0 = {}, c1 = {};
      mm_acc2(c0, c1, s_y, 32,  Wag + OFF_DEC_WY, 32,  nt0, nt1, 1, lane);
      mm_acc2(c0, c1, s_z, 64,  Wag + OFF_DEC_WZ, 64,  nt0, nt1, 2, lane);
      mm_acc2(c0, c1, s_h, 256, Wag + OFF_DEC_WH, 256, nt0, nt1, 8, lane);
      epi_gather_relu(c0, nt0, g, b1, s_idx, s_a1, mb, nl);
      epi_gather_relu(c1, nt1, g, b1, s_idx, s_a1, mb, nl);
    }
    __syncthreads();
    layer2_relu(s_a1, Wag + OFF_DEC_W2, p.decB2 + a * 256, s_a2, wave, lane, mb, nl);
    __syncthreads();
    if (wave == 0) head2(s_a2, Wag + OFF_DEC_WMH, p.decmB + a * 2, s_dm, lane, mb, nl);
    if (wave == 1) head2(s_a2, Wag + OFF_DEC_WSH, p.decsB + a * 2, s_ds, lane, mb, nl);
    __syncthreads();

    // ---- recon accumulation (16 x 2 elements) ----
    if (tid < 32) {
      int m = tid >> 1, j = tid & 1;
      float mean = s_dm[m * 16 + j];
      float sd   = softplusf(s_ds[m * 16 + j]);
      float d    = (s_xf[m * 2 + j] - mean) / sd;
      rc_acc += 0.5f * d * d + logf(sd) + 0.5f * LOG2PI_;
    }

    // ================= GRU =================
    {
      const unsigned short* Wgx = Wag + OFF_GRU_WX;
      const unsigned short* Wgz = Wag + OFF_GRU_WZ;
      const unsigned short* Wgh = Wag + OFF_GRU_WH;
      const float* bih = p.gruBih + a * 768;
      const float* bhh = p.gruBhh + a * 768;
      #pragma unroll
      for (int ci = 0; ci < 2; ++ci) {
        int cc = wave * 2 + ci;                 // h column tile 0..15
        f32x8 gir = {}, giu = {}, gin = {}, ghr = {}, ghu = {}, ghn = {};
        mm_acc3(gir, giu, gin, s_x, 32,  Wgx, 32,  cc, 16 + cc, 32 + cc, 1, lane);
        mm_acc3(gir, giu, gin, s_z, 64,  Wgz, 64,  cc, 16 + cc, 32 + cc, 2, lane);
        mm_acc3(ghr, ghu, ghn, s_h, 256, Wgh, 256, cc, 16 + cc, 32 + cc, 8, lane);
        int n0 = cc * 16 + nl;
        float bir = ldg_f(bih + n0),       bhr = ldg_f(bhh + n0);
        float biu = ldg_f(bih + 256 + n0), bhu = ldg_f(bhh + 256 + n0);
        float bin = ldg_f(bih + 512 + n0), bhn = ldg_f(bhh + 512 + n0);
        #pragma unroll
        for (int v = 0; v < 8; ++v) {
          float r  = sigmoidf_(gir[v] + bir + ghr[v] + bhr);
          float u  = sigmoidf_(giu[v] + biu + ghu[v] + bhu);
          float nn = tanhf(gin[v] + bin + r * (ghn[v] + bhn));
          hreg[ci][v] = (1.f - u) * nn + u * hreg[ci][v];
        }
      }
    }
    __syncthreads();      // everyone done reading old h
    #pragma unroll
    for (int ci = 0; ci < 2; ++ci) {
      int cc = wave * 2 + ci;
      #pragma unroll
      for (int v = 0; v < 8; ++v)
        s_h[(mb + v) * 256 + cc * 16 + nl] = f2bf(hreg[ci][v]);
    }
    __syncthreads();
  }

  // ---- reduce & accumulate into d_out ----
  atomicAdd(&s_red[0], rc_acc);
  atomicAdd(&s_red[1], kl_acc);
  __syncthreads();
  if (tid == 0) {
    atomicAdd(p.out + 0, s_red[0]);
    atomicAdd(p.out + 1, s_red[1]);
  }
}

// ---- prep: W[a,row0+k,n] (f32) -> dst[a][n][k] (bf16, zero-padded) ----
__global__ void prepT(const float* src, unsigned short* dst,
                      long long srcAstride, int row0, int srcN,
                      int Nreal, int Npad, int Kreal, int Kpad,
                      long long dstOff, long long dstAstride) {
  int a = blockIdx.y;
  int i = blockIdx.x * 256 + threadIdx.x;
  if (i >= Npad * Kpad) return;
  int n = i / Kpad, k = i % Kpad;
  float v = 0.f;
  if (n < Nreal && k < Kreal)
    v = ldg_f(src + a * srcAstride + (long long)(row0 + k) * srcN + n);
  dst[a * dstAstride + dstOff + (long long)n * Kpad + k] = f2bf(v);
}

__global__ void zero_out_k(float* o) {
  if (threadIdx.x < 2) o[threadIdx.x] = 0.f;
}

extern "C" void kernel_launch(void* const* d_in, const int* in_sizes, int n_in,
                              void* d_out, int out_size, void* d_ws, size_t ws_size,
                              hipStream_t stream) {
  (void)in_sizes; (void)n_in; (void)out_size; (void)ws_size;
  unsigned short* W = (unsigned short*)d_ws;
  float* out = (float*)d_out;

  zero_out_k<<<1, 32, 0, stream>>>(out);

  auto prep = [&](const void* src, long long sA, int row0, int srcN,
                  int Nreal, int Npad, int Kreal, int Kpad, long long off) {
    int total = Npad * Kpad;
    dim3 g((total + 255) / 256, A_);
    prepT<<<g, 256, 0, stream>>>((const float*)src, W, sA, row0, srcN,
                                 Nreal, Npad, Kreal, Kpad, off, AGW_);
  };

  // encoder
  prep(d_in[3],  348LL * 256, 0,   256, 256, 256, 2,   32,  OFF_ENC_WX);
  prep(d_in[3],  348LL * 256, 92,  256, 256, 256, 256, 256, OFF_ENC_WH);
  prep(d_in[5],  256LL * 256, 0,   256, 256, 256, 256, 256, OFF_ENC_W2);
  prep(d_in[7],  256LL * 64,  0,   64,  64,  64,  256, 256, OFF_ENC_WM);
  prep(d_in[9],  256LL * 64,  0,   64,  64,  64,  256, 256, OFF_ENC_WS);
  // prior
  prep(d_in[11], 346LL * 256, 90,  256, 256, 256, 256, 256, OFF_PRI_WH);
  prep(d_in[13], 256LL * 256, 0,   256, 256, 256, 256, 256, OFF_PRI_W2);
  prep(d_in[15], 256LL * 64,  0,   64,  64,  64,  256, 256, OFF_PRI_WM);
  prep(d_in[17], 256LL * 64,  0,   64,  64,  64,  256, 256, OFF_PRI_WS);
  // decoder
  prep(d_in[19], 430LL * 256, 0,   256, 256, 256, 20,  32,  OFF_DEC_WY);
  prep(d_in[19], 430LL * 256, 110, 256, 256, 256, 64,  64,  OFF_DEC_WZ);
  prep(d_in[19], 430LL * 256, 174, 256, 256, 256, 256, 256, OFF_DEC_WH);
  prep(d_in[21], 256LL * 256, 0,   256, 256, 256, 256, 256, OFF_DEC_W2);
  prep(d_in[23], 256LL * 2,   0,   2,   2,   16,  256, 256, OFF_DEC_WMH);
  prep(d_in[25], 256LL * 2,   0,   2,   2,   16,  256, 256, OFF_DEC_WSH);
  // GRU
  prep(d_in[27], 66LL * 768,  0,   768, 768, 768, 2,   32,  OFF_GRU_WX);
  prep(d_in[27], 66LL * 768,  2,   768, 768, 768, 64,  64,  OFF_GRU_WZ);
  prep(d_in[29], 256LL * 768, 0,   768, 768, 768, 256, 256, OFF_GRU_WH);

  Params p;
  p.data  = (const float*)d_in[0];
  p.mac   = (const int*)  d_in[1];
  p.eps   = (const float*)d_in[2];
  p.encW1 = (const float*)d_in[3];  p.encB1 = (const float*)d_in[4];
  p.encB2 = (const float*)d_in[6];
  p.encmB = (const float*)d_in[8];  p.encsB = (const float*)d_in[10];
  p.priW1 = (const float*)d_in[11]; p.priB1 = (const float*)d_in[12];
  p.priB2 = (const float*)d_in[14];
  p.primB = (const float*)d_in[16]; p.prisB = (const float*)d_in[18];
  p.decW1 = (const float*)d_in[19]; p.decB1 = (const float*)d_in[20];
  p.decB2 = (const float*)d_in[22];
  p.decmB = (const float*)d_in[24]; p.decsB = (const float*)d_in[26];
  p.gruBih = (const float*)d_in[28];
  p.gruBhh = (const float*)d_in[30];
  p.W = W; p.out = out;

  vrnn_persistent<<<dim3(B_ / 16, A_), 256, 0, stream>>>(p);
}
